// MultiAttention_18957985645095
// MI455X (gfx1250) — matrix operand order repaired and checked by reading, not compiled, not run on hardware
//
#include <hip/hip_runtime.h>

// ---------------------------------------------------------------------------
// Multi-head attention forward for MI455X (gfx1250).
// bf16 WMMA (16x16x32) + Tensor Data Mover (TDM) async tile loads.
// d_out = [ out (2*2048*1024 f32) | attn (2*16*2048*2048 f32) ]
// d_ws  = [ q bf16 8MB | k bf16 8MB | vT bf16 8MB | ctx bf16 8MB ]
// ---------------------------------------------------------------------------

typedef __attribute__((ext_vector_type(16))) __bf16 v16bf;
typedef __attribute__((ext_vector_type(8)))  float  v8f;
typedef __attribute__((ext_vector_type(4)))  unsigned int v4u;
typedef __attribute__((ext_vector_type(8)))  int    v8i;
typedef __attribute__((ext_vector_type(4)))  int    v4i;

namespace cfg {
constexpr int S     = 2048;   // sequence length
constexpr int HD    = 64;     // head dim
constexpr int NH    = 16;     // heads
constexpr int MDIM  = 1024;   // model dim
constexpr int MROWS = 4096;   // B*S
}

// --- WMMA operand builders (ISA 7.12.2 layouts, wave32) ---------------------
// A (16x32 bf16): lanes 0-15 row M=lane hold K=0..7 then K=16..23;
//                 lanes 16-31 same row hold K=8..15 then K=24..31.
__device__ __forceinline__ v16bf load_opA(const __bf16* tile, int row,
                                          int stride, int colbase, int lane) {
  const __bf16* p = tile + row * stride + colbase + ((lane & 16) ? 8 : 0);
  v16bf r;
#pragma unroll
  for (int j = 0; j < 8; ++j) { r[j] = p[j]; r[j + 8] = p[j + 16]; }
  return r;
}
// B (32x16 bf16): lane L<16 holds column N=L, K=0..7 then K=16..23;
//                 lane L+16 holds column N=L, K=8..15 then K=24..31. Tile staged [N][K].
__device__ __forceinline__ v16bf load_opB(const __bf16* tile, int row,
                                          int stride, int colbase, int lane) {
  const __bf16* p = tile + row * stride + colbase + ((lane & 16) ? 8 : 0);
  v16bf r;
#pragma unroll
  for (int j = 0; j < 8; ++j) { r[j] = p[j]; r[j + 8] = p[j + 16]; }
  return r;
}

// --- Tensor Data Mover: 2D bf16 tile -> LDS (D# per ISA 8.3/8.4) ------------
// pad_enable reproduces padded LDS row stride: pad_amt_code+1 DWORDs appended
// every 2<<pad_int_code DWORDs.
__device__ __forceinline__ void tdm_load_tile_bf16(
    const __bf16* src, void* lds_dst, unsigned tensor_d0, unsigned tensor_d1,
    unsigned long long stride0, unsigned tile_d0, unsigned tile_d1,
    unsigned pad_int_code, unsigned pad_amt_code) {
  const unsigned long long ga = (unsigned long long)(const void*)src;
  const unsigned lds = (unsigned)(unsigned long long)lds_dst;  // LDS byte offset
  v4u g0;
  g0[0] = 1u;                                   // count=1 (valid), user mode
  g0[1] = lds;                                  // lds_addr
  g0[2] = (unsigned)ga;                         // global_addr[31:0]
  g0[3] = (unsigned)((ga >> 32) & 0x01FFFFFFu) | (2u << 30);  // [56:32], type=2
  v8i g1;
  g1[0] = (int)((1u << 16)                      // data_size = 1 (2 bytes)
                | (1u << 20)                    // pad_enable
                | (pad_int_code << 22) | (pad_amt_code << 25));
  g1[1] = (int)((tensor_d0 & 0xFFFFu) << 16);                    // dim0[15:0]
  g1[2] = (int)(((tensor_d0 >> 16) & 0xFFFFu) |
                ((tensor_d1 & 0xFFFFu) << 16));                  // dim0/dim1
  g1[3] = (int)(((tensor_d1 >> 16) & 0xFFFFu) | (tile_d0 << 16));
  g1[4] = (int)(tile_d1 & 0xFFFFu);            // tile_dim1; tile_dim2 = 0
  g1[5] = (int)(unsigned)(stride0 & 0xFFFFFFFFull);
  g1[6] = (int)(unsigned)((stride0 >> 32) & 0xFFFFull);  // dim1_stride = 0 (2D)
  g1[7] = 0;
  const v4i z4 = {0, 0, 0, 0};
#if __clang_major__ >= 23
  const v8i z8 = {0, 0, 0, 0, 0, 0, 0, 0};
  __builtin_amdgcn_tensor_load_to_lds(g0, g1, z4, z4, z8, 0);
#else
  __builtin_amdgcn_tensor_load_to_lds(g0, g1, z4, z4, 0);
#endif
}

// ---------------------------------------------------------------------------
// Projection GEMM: D[m,n] = sum_k A[m,k]*W[n,k] + bias[n]; K = 1024.
// Block 256 thr (8 waves, 2x4). Block tile 64(M) x 128(N), BK=32.
// Each wave owns a 32x32 tile -> 4 WMMA per K step, 2x operand reuse.
// MODE 0: out bf16 [B,H,S,D] (q,k)  MODE 1: bf16 [B,H,D,S] (vT)
// MODE 2: out f32 [M,N] (final projection)
// ---------------------------------------------------------------------------
template <int MODE, typename AT>
__global__ __launch_bounds__(256) void gemm_bias_kernel(
    const AT* __restrict__ A, const float* __restrict__ W,
    const float* __restrict__ bias, void* __restrict__ outp) {
  __shared__ __align__(16) __bf16 As[64 * 40];
  __shared__ __align__(16) __bf16 Bs[128 * 40];

  const int tid  = threadIdx.x;
  const int lane = tid & 31, wave = tid >> 5;
  const int wm = wave >> 2, wn = wave & 3;
  const int mBlk = blockIdx.y * 64;
  const int nBlk = blockIdx.x * 128;
  const int ra = tid >> 2, ca = (tid & 3) << 3;   // A: 64x32, 8 elems/thread
  const int rb = tid >> 1, cb = (tid & 1) << 4;   // W: 128x32, 16 elems/thread

  v8f acc[2][2] = {};

  for (int kk = 0; kk < cfg::MDIM; kk += 32) {
    const AT* ga = A + (size_t)(mBlk + ra) * cfg::MDIM + kk + ca;
#pragma unroll
    for (int j = 0; j < 8; ++j) As[ra * 40 + ca + j] = (__bf16)ga[j];
    const float* gw = W + (size_t)(nBlk + rb) * cfg::MDIM + kk + cb;
#pragma unroll
    for (int j = 0; j < 16; ++j) Bs[rb * 40 + cb + j] = (__bf16)gw[j];
    if (kk + 32 < cfg::MDIM) {      // gfx1250 global_prefetch_b8 of next tiles
      __builtin_prefetch(ga + 32, 0, 0);
      __builtin_prefetch(gw + 32, 0, 0);
    }
    __syncthreads();

    v16bf aop[2], bop[2];
#pragma unroll
    for (int mi = 0; mi < 2; ++mi)
      aop[mi] = load_opA(As, wm * 32 + mi * 16 + (lane & 15), 40, 0, lane);
#pragma unroll
    for (int ni = 0; ni < 2; ++ni)
      bop[ni] = load_opB(Bs, wn * 32 + ni * 16 + (lane & 15), 40, 0, lane);
#pragma unroll
    for (int mi = 0; mi < 2; ++mi)
#pragma unroll
      for (int ni = 0; ni < 2; ++ni)
        acc[mi][ni] = __builtin_amdgcn_wmma_f32_16x16x32_bf16(
            false, aop[mi], false, bop[ni], (short)0, acc[mi][ni], false, false);
    __syncthreads();
  }

  // Epilogue: lanes 0-15: N=lane, VGPR r -> M=r; lanes 16-31: M=8+r.
#pragma unroll
  for (int mi = 0; mi < 2; ++mi) {
#pragma unroll
    for (int ni = 0; ni < 2; ++ni) {
      const int n  = nBlk + wn * 32 + ni * 16 + (lane & 15);
      const int mb = mBlk + wm * 32 + mi * 16 + ((lane >> 4) << 3);
      const float bval = bias[n];
#pragma unroll
      for (int r2 = 0; r2 < 8; ++r2) {
        const int m = mb + r2;
        const float val = acc[mi][ni][r2] + bval;
        if (MODE == 0) {
          const int b = m >> 11, s = m & 2047, h = n >> 6, d = n & 63;
          ((__bf16*)outp)[(((size_t)(b * cfg::NH + h)) * cfg::S + s) * cfg::HD +
                          d] = (__bf16)val;
        } else if (MODE == 1) {
          const int b = m >> 11, s = m & 2047, h = n >> 6, d = n & 63;
          ((__bf16*)outp)[(((size_t)(b * cfg::NH + h)) * cfg::HD + d) * cfg::S +
                          s] = (__bf16)val;
        } else {
          ((float*)outp)[(size_t)m * cfg::MDIM + n] = val;
        }
      }
    }
  }
}

// ---------------------------------------------------------------------------
// scores[bh,sq,sk] = (1/8) * sum_d q[bh,sq,d]*k[bh,sk,d]
// Block tile 64(sq) x 128(sk). Whole D=64 strips of Q (64x64) and K (128x64)
// are loaded once by the TDM; then 8 WMMA/wave with no further staging.
// ---------------------------------------------------------------------------
__global__ __launch_bounds__(256) void scores_kernel(
    const __bf16* __restrict__ q, const __bf16* __restrict__ k,
    float* __restrict__ attn) {
  const int bh  = blockIdx.z;
  const int sq0 = blockIdx.y * 64;
  const int sk0 = blockIdx.x * 128;
  if (sk0 > sq0 + 63) return;  // fully above causal diagonal

  __shared__ __align__(16) __bf16 Qs[64 * 72];    // 64 rows x 64 cols, pad 8
  __shared__ __align__(16) __bf16 Ks[128 * 72];

  const int tid  = threadIdx.x;
  const int lane = tid & 31, wave = tid >> 5;
  const int wm = wave >> 2, wn = wave & 3;

  const __bf16* qb = q + (size_t)bh * cfg::S * cfg::HD;
  const __bf16* kb = k + (size_t)bh * cfg::S * cfg::HD;

  if (wave == 0) {  // TDM ignores EXEC: issue once per block
    // rows of 64 bf16 = 32 DWORDs, +4 DWORD pad -> LDS stride 72 elems
    tdm_load_tile_bf16(qb + (size_t)sq0 * cfg::HD, Qs, cfg::HD, cfg::S, cfg::HD,
                       64, 64, /*interval 32dw*/ 4, /*pad 4dw*/ 3);
    tdm_load_tile_bf16(kb + (size_t)sk0 * cfg::HD, Ks, cfg::HD, cfg::S, cfg::HD,
                       64, 128, 4, 3);
    __builtin_amdgcn_s_wait_tensorcnt((short)0);
  }
  __syncthreads();

  v8f acc[2][2] = {};
#pragma unroll
  for (int kk = 0; kk < cfg::HD; kk += 32) {
    v16bf aop[2], bop[2];
#pragma unroll
    for (int mi = 0; mi < 2; ++mi)
      aop[mi] = load_opA(Qs, wm * 32 + mi * 16 + (lane & 15), 72, kk, lane);
#pragma unroll
    for (int ni = 0; ni < 2; ++ni)
      bop[ni] = load_opB(Ks, wn * 32 + ni * 16 + (lane & 15), 72, kk, lane);
#pragma unroll
    for (int mi = 0; mi < 2; ++mi)
#pragma unroll
      for (int ni = 0; ni < 2; ++ni)
        acc[mi][ni] = __builtin_amdgcn_wmma_f32_16x16x32_bf16(
            false, aop[mi], false, bop[ni], (short)0, acc[mi][ni], false, false);
  }

  float* arow = attn + (size_t)bh * cfg::S * cfg::S;
#pragma unroll
  for (int mi = 0; mi < 2; ++mi) {
#pragma unroll
    for (int ni = 0; ni < 2; ++ni) {
      const int n  = sk0 + wn * 32 + ni * 16 + (lane & 15);
      const int mb = sq0 + wm * 32 + mi * 16 + ((lane >> 4) << 3);
#pragma unroll
      for (int r2 = 0; r2 < 8; ++r2)
        arow[(size_t)(mb + r2) * cfg::S + n] = acc[mi][ni][r2] * 0.125f;
    }
  }
}

// ---------------------------------------------------------------------------
// In-place causal softmax; one block per (bh,sq) row, tail written as 0.
// ---------------------------------------------------------------------------
__global__ __launch_bounds__(256) void softmax_kernel(float* __restrict__ attn) {
  const int row = blockIdx.x;
  const int sq  = row & (cfg::S - 1);
  const int valid = sq + 1;
  float* p = attn + (size_t)row * cfg::S;
  const int tid = threadIdx.x;
  __shared__ float red[256];

  float mx = -INFINITY;
  for (int j = tid; j < valid; j += 256) mx = fmaxf(mx, p[j]);
  red[tid] = mx;
  __syncthreads();
  for (int s = 128; s > 0; s >>= 1) {
    if (tid < s) red[tid] = fmaxf(red[tid], red[tid + s]);
    __syncthreads();
  }
  mx = red[0];
  __syncthreads();

  float sum = 0.f;
  for (int j = tid; j < valid; j += 256) sum += __expf(p[j] - mx);
  red[tid] = sum;
  __syncthreads();
  for (int s = 128; s > 0; s >>= 1) {
    if (tid < s) red[tid] += red[tid + s];
    __syncthreads();
  }
  const float inv = 1.f / red[0];

  for (int j = tid; j < cfg::S; j += 256)
    p[j] = (j < valid) ? __expf(p[j] - mx) * inv : 0.f;
}

// ---------------------------------------------------------------------------
// ctx[b,sq,h*64+d] = sum_sk attn[bh,sq,sk] * v[bh,sk,d]
// Block tile 64(sq) x 64(d); vT tile via TDM each K step while threads convert
// the f32 attn tile. K loop truncated at the causal boundary.
// ---------------------------------------------------------------------------
__global__ __launch_bounds__(256) void attnv_kernel(
    const float* __restrict__ attn, const __bf16* __restrict__ vT,
    __bf16* __restrict__ ctx) {
  const int bh  = blockIdx.z;
  const int sq0 = blockIdx.y * 64;

  __shared__ __align__(16) __bf16 As[64 * 40];
  __shared__ __align__(16) __bf16 Bs[64 * 40];

  const int tid  = threadIdx.x;
  const int lane = tid & 31, wave = tid >> 5;
  const int wm = wave >> 2, wn = wave & 3;   // wave tile: 32(sq) x 16(d)
  const int ra = tid >> 2, ca = (tid & 3) << 3;

  const float*  ab = attn + (size_t)bh * cfg::S * cfg::S;
  const __bf16* vb = vT + (size_t)bh * cfg::HD * cfg::S;

  v8f acc[2] = {};
  const int kmax = sq0 + 64;  // rows sq0..sq0+63 are zero beyond column sq
  for (int kk = 0; kk < kmax; kk += 32) {
    if (wave == 0)  // vT tile: 64 rows(d) x 32 cols(sk), stride S
      tdm_load_tile_bf16(vb + kk, Bs, cfg::S, cfg::HD, cfg::S, 32, 64,
                         /*interval 16dw*/ 3, /*pad 4dw*/ 3);
    const float* ga = ab + (size_t)(sq0 + ra) * cfg::S + kk + ca;
#pragma unroll
    for (int j = 0; j < 8; ++j) As[ra * 40 + ca + j] = (__bf16)ga[j];
    if (wave == 0) __builtin_amdgcn_s_wait_tensorcnt((short)0);
    __syncthreads();

    v16bf bop = load_opB(Bs, wn * 16 + (lane & 15), 40, 0, lane);
#pragma unroll
    for (int mi = 0; mi < 2; ++mi) {
      v16bf aop = load_opA(As, wm * 32 + mi * 16 + (lane & 15), 40, 0, lane);
      acc[mi] = __builtin_amdgcn_wmma_f32_16x16x32_bf16(
          false, aop, false, bop, (short)0, acc[mi], false, false);
    }
    __syncthreads();
  }

  const int n = wn * 16 + (lane & 15);  // d
  const int b = bh >> 4, h = bh & 15;
#pragma unroll
  for (int mi = 0; mi < 2; ++mi) {
    const int mb = sq0 + wm * 32 + mi * 16 + ((lane >> 4) << 3);
#pragma unroll
    for (int r2 = 0; r2 < 8; ++r2)
      ctx[((size_t)(b * cfg::S + mb + r2)) * cfg::MDIM + h * cfg::HD + n] =
          (__bf16)acc[mi][r2];
  }
}

// ---------------------------------------------------------------------------
extern "C" void kernel_launch(void* const* d_in, const int* in_sizes, int n_in,
                              void* d_out, int out_size, void* d_ws,
                              size_t ws_size, hipStream_t stream) {
  (void)in_sizes; (void)n_in; (void)out_size; (void)ws_size;
  const float* key   = (const float*)d_in[0];
  const float* value = (const float*)d_in[1];
  const float* query = (const float*)d_in[2];
  // d_in[3] = causal mask (structure applied analytically; unused)
  const float* Wk = (const float*)d_in[4];
  const float* bk = (const float*)d_in[5];
  const float* Wv = (const float*)d_in[6];
  const float* bv = (const float*)d_in[7];
  const float* Wq = (const float*)d_in[8];
  const float* bq = (const float*)d_in[9];
  const float* Wo = (const float*)d_in[10];
  const float* bo = (const float*)d_in[11];

  float* out  = (float*)d_out;
  float* attn = out + (size_t)cfg::MROWS * cfg::MDIM;

  __bf16* qws = (__bf16*)d_ws;
  __bf16* kws = qws + (size_t)cfg::MROWS * cfg::MDIM;
  __bf16* vws = kws + (size_t)cfg::MROWS * cfg::MDIM;
  __bf16* ctx = vws + (size_t)cfg::MROWS * cfg::MDIM;

  const dim3 blk(256);
  const dim3 gproj(cfg::MDIM / 128, cfg::MROWS / 64);       // 8 x 64
  const dim3 gsc(cfg::S / 128, cfg::S / 64, 2 * cfg::NH);   // 16 x 32 x 32
  const dim3 gsm(2 * cfg::NH * cfg::S);                     // 65536
  const dim3 gav(1, cfg::S / 64, 2 * cfg::NH);              // 1 x 32 x 32

  gemm_bias_kernel<0, float><<<gproj, blk, 0, stream>>>(query, Wq, bq, qws);
  gemm_bias_kernel<0, float><<<gproj, blk, 0, stream>>>(key,   Wk, bk, kws);
  gemm_bias_kernel<1, float><<<gproj, blk, 0, stream>>>(value, Wv, bv, vws);
  scores_kernel<<<gsc, blk, 0, stream>>>(qws, kws, attn);
  softmax_kernel<<<gsm, blk, 0, stream>>>(attn);
  attnv_kernel<<<gav, blk, 0, stream>>>(attn, vws, ctx);
  gemm_bias_kernel<2, __bf16><<<gproj, blk, 0, stream>>>(ctx, Wo, bo, out);
}
